// Model_59030030516227
// MI455X (gfx1250) — matrix operand order repaired and checked by reading, not compiled, not run on hardware
//
#include <hip/hip_runtime.h>
#include <hip/hip_bf16.h>

// ---------------- problem constants (match reference) ----------------
#define NN 19621
#define EE 627872
#define BB 16
#define TT 4
#define HOPS 75      // demand hops -> 76 features incl. hop 0
#define FEAT 76
#define SFEAT 15     // status-diffusion features (1+2+4+8)

typedef __attribute__((ext_vector_type(16))) _Float16 v16h;
typedef __attribute__((ext_vector_type(8)))  _Float16 v8h;
typedef __attribute__((ext_vector_type(8)))  float    v8f;

// ---------------------------------------------------------------------
// CSR construction (on-device, once per launch)
// ---------------------------------------------------------------------
__global__ void k_zero_i(int* __restrict__ p, int n) {
    int i = blockIdx.x * 256 + threadIdx.x;
    if (i < n) p[i] = 0;
}

__global__ void k_hist(const int* __restrict__ src, const int* __restrict__ dst,
                       int* __restrict__ cntA, int* __restrict__ cntB, int E) {
    int e = blockIdx.x * 256 + threadIdx.x;
    if (e >= E) return;
    atomicAdd(&cntA[src[e]], 1);   // group by src  (CSR_A : rows = src)
    atomicAdd(&cntB[dst[e]], 1);   // group by dst  (CSR_B : rows = dst)
}

// single-block exclusive scan over N counts (LDS Hillis-Steele over 1024 partials)
__global__ __launch_bounds__(1024) void k_scan(const int* __restrict__ cnt,
                                               int* __restrict__ rowptr, int N) {
    __shared__ int part[1024];
    const int t = threadIdx.x;
    const int chunk = (N + 1023) / 1024;
    const int begin = t * chunk;
    const int end   = min(begin + chunk, N);
    int s = 0;
    for (int i = begin; i < end; ++i) s += cnt[i];
    part[t] = s;
    __syncthreads();
    for (int off = 1; off < 1024; off <<= 1) {
        int v = (t >= off) ? part[t - off] : 0;
        __syncthreads();
        part[t] += v;
        __syncthreads();
    }
    int excl = (t == 0) ? 0 : part[t - 1];
    for (int i = begin; i < end; ++i) { rowptr[i] = excl; excl += cnt[i]; }
    if (t == 1023) rowptr[N] = part[1023];
}

__global__ void k_copy_i(const int* __restrict__ a, int* __restrict__ b, int n) {
    int i = blockIdx.x * 256 + threadIdx.x;
    if (i < n) b[i] = a[i];
}

__global__ void k_scatter(const int* __restrict__ src, const int* __restrict__ dst,
                          const float* __restrict__ w_vals, const float* __restrict__ t_vals,
                          int* __restrict__ curA, int* __restrict__ curB,
                          int* __restrict__ nbrA, float* __restrict__ wA, float* __restrict__ tA,
                          int* __restrict__ nbrB, float* __restrict__ wB, int E) {
    int e = blockIdx.x * 256 + threadIdx.x;
    if (e >= E) return;
    const int s = src[e], d = dst[e];
    // CSR_A (rows = src): used for y[src] += v * x[dst]  (A^T hops, W^T)
    int pA = atomicAdd(&curA[s], 1);
    nbrA[pA] = d;
    wA[pA]   = w_vals[e];
    #pragma unroll
    for (int t = 0; t < TT; ++t) tA[(size_t)t * E + pA] = t_vals[(size_t)t * E + e];
    // CSR_B (rows = dst): used for y[dst] += v * x[src]  (W)
    int pB = atomicAdd(&curB[d], 1);
    nbrB[pB] = s;
    wB[pB]   = w_vals[e];
}

// ---------------------------------------------------------------------
// diffusion: gather-form SpMV, 16 batch lanes per node (64B coalesced gathers)
// ---------------------------------------------------------------------
__global__ __launch_bounds__(256) void k_spmv(const int* __restrict__ rowptr,
                                              const int* __restrict__ nbr,
                                              const float* __restrict__ vals,
                                              const float* __restrict__ xin,   // [N][16]
                                              float* __restrict__ yout,        // [N][16]
                                              int N) {
    const int b = threadIdx.x & 15;
    const int n = blockIdx.x * 16 + (threadIdx.x >> 4);
    if (n >= N) return;
    const int beg = rowptr[n], end = rowptr[n + 1];
    float acc = 0.f;
    for (int j = beg; j < end; ++j)
        acc = fmaf(vals[j], xin[(size_t)nbr[j] * 16 + b], acc);
    yout[(size_t)n * 16 + b] = acc;
}

__global__ void k_transpose_x(const float* __restrict__ x,  // [B][T][N]
                              float* __restrict__ h0, float* __restrict__ s0,
                              int t, int N) {
    int gid = blockIdx.x * 256 + threadIdx.x;
    if (gid >= BB * N) return;
    int b = gid / N, n = gid % N;
    float v = x[((size_t)b * TT + t) * N + n];
    h0[(size_t)n * 16 + b] = v;
    s0[(size_t)n * 16 + b] = v;
}

// ---------------------------------------------------------------------
// per-node attention logits via WMMA:  L[b][o] = S_n[b][:15] @ att_w[n] + att_b[n]
// one wave per node (4 nodes/block). Operands staged zero-padded in LDS so
// fragment builds are UNconditional ds_load_b128 pairs (no EXEC juggling).
//   A: 16x32 f16 (K padded 15->32)   sA[node][b][K=32]
//   B: 32x16 f16 per 16-wide o tile  sW[node][o=80][K=32] (K>=15 zeros)
// ---------------------------------------------------------------------
__global__ __launch_bounds__(128) void k_wmma_att(const float* __restrict__ S,      // [15][N][16]
                                                  const float* __restrict__ att_w,  // [N][15][76]
                                                  const float* __restrict__ att_b,  // [N][76]
                                                  float* __restrict__ L,            // [76][N][16]
                                                  int N) {
    __shared__ alignas(16) _Float16 sA[4][16][32];   //  4 KB
    __shared__ alignas(16) _Float16 sW[4][80][32];   // 20 KB

    const int tid = threadIdx.x;
    const int n0  = blockIdx.x * 4;

    // ---- zero-fill both staging buffers (covers all padding) ----
    {
        int* zA = (int*)&sA[0][0][0];                       // 1024 dwords
        int* zW = (int*)&sW[0][0][0];                       // 5120 dwords
        for (int i = tid; i < 1024; i += 128) zA[i] = 0;
        for (int i = tid; i < 5120; i += 128) zW[i] = 0;
    }
    __syncthreads();

    // ---- stage S (coalesced over b) ----
    for (int idx = tid; idx < 4 * SFEAT * 16; idx += 128) {
        const int nl  = idx / (SFEAT * 16);
        const int rem = idx - nl * (SFEAT * 16);
        const int K   = rem >> 4;
        const int b   = rem & 15;
        const int n   = n0 + nl;
        if (n < N) sA[nl][b][K] = (_Float16)S[((size_t)K * N + n) * 16 + b];
    }
    // ---- stage att_w (coalesced over o) ----
    for (int idx = tid; idx < 4 * SFEAT * FEAT; idx += 128) {
        const int nl  = idx / (SFEAT * FEAT);
        const int rem = idx - nl * (SFEAT * FEAT);
        const int K   = rem / FEAT;
        const int o   = rem - K * FEAT;
        const int n   = n0 + nl;
        if (n < N) sW[nl][o][K] = (_Float16)att_w[(size_t)n * (SFEAT * FEAT) + K * FEAT + o];
    }
    __syncthreads();

    const int wave = tid >> 5;
    const int lane = tid & 31;
    const int n = n0 + wave;
    if (n >= N) return;                   // wave-uniform; no further barriers

    const int g   = lane >> 4;            // lane group
    const int sub = lane & 15;            // A: batch row b ; B/D: column in tile

    // ---- A fragment: lane holds K = [8g, 8g+8) and [16+8g, 16+8g+8) of row b=sub ----
    const v8h aLo = *(const v8h*)&sA[wave][sub][8 * g];
    const v8h aHi = *(const v8h*)&sA[wave][sub][16 + 8 * g];
    v16h a;
    #pragma unroll
    for (int i = 0; i < 8; ++i) { a[i] = aLo[i]; a[8 + i] = aHi[i]; }

    const float* bn = att_b + (size_t)n * FEAT;

    #pragma unroll
    for (int tile = 0; tile < 5; ++tile) {
        const int o = tile * 16 + sub;    // o < 80 always; sW padded to 80 rows
        // ---- B fragment: K = [8g, 8g+8) and [16+8g, 16+8g+8) of column o (zeros for K>=15) ----
        const v8h w0 = *(const v8h*)&sW[wave][o][8 * g];
        const v8h w1 = *(const v8h*)&sW[wave][o][16 + 8 * g];
        v16h bf;
        #pragma unroll
        for (int i = 0; i < 8; ++i) { bf[i] = w0[i]; bf[8 + i] = w1[i]; }

        v8f c = {};
        c = __builtin_amdgcn_wmma_f32_16x16x32_f16(false, a, false, bf,
                                                   (short)0, c, false, false);
        if (o < FEAT) {                   // divergent only for tile 4, after the WMMA
            const float bias = bn[o];
            float* dst = &L[((size_t)o * N + n) * 16 + 8 * g];  // rows b = 8g..8g+7
            float4 s0 = make_float4(c[0] + bias, c[1] + bias, c[2] + bias, c[3] + bias);
            float4 s1 = make_float4(c[4] + bias, c[5] + bias, c[6] + bias, c[7] + bias);
            *(float4*)(dst)     = s0;
            *(float4*)(dst + 4) = s1;
        }
    }
}

// ---------------------------------------------------------------------
// deterministic softmax over nodes (dim=2): two-stage tree reductions
// ---------------------------------------------------------------------
__global__ __launch_bounds__(256) void k_maxpart(const float* __restrict__ L,
                                                 float* __restrict__ P,  // [76][16][16]
                                                 int N) {
    const int o = blockIdx.x, slice = blockIdx.y;
    const int b = threadIdx.x & 15, nl = threadIdx.x >> 4;
    float m = -3.4e38f;
    for (int n = slice * 16 + nl; n < N; n += 256)
        m = fmaxf(m, L[((size_t)o * N + n) * 16 + b]);
    __shared__ float red[256];
    red[threadIdx.x] = m;
    __syncthreads();
    for (int s = 128; s >= 16; s >>= 1) {
        if (threadIdx.x < s) red[threadIdx.x] = fmaxf(red[threadIdx.x], red[threadIdx.x + s]);
        __syncthreads();
    }
    if (threadIdx.x < 16) P[((size_t)o * 16 + slice) * 16 + b] = red[threadIdx.x];
}

__global__ void k_maxfin(const float* __restrict__ P, float* __restrict__ M) {
    const int o = blockIdx.x, b = threadIdx.x;
    if (b >= 16) return;
    float m = -3.4e38f;
    for (int s = 0; s < 16; ++s) m = fmaxf(m, P[((size_t)o * 16 + s) * 16 + b]);
    M[(size_t)o * 16 + b] = m;
}

__global__ __launch_bounds__(256) void k_expsum(float* __restrict__ L,
                                                const float* __restrict__ M,
                                                float* __restrict__ P, int N) {
    const int o = blockIdx.x, slice = blockIdx.y;
    const int b = threadIdx.x & 15, nl = threadIdx.x >> 4;
    const float m = M[(size_t)o * 16 + b];
    float s = 0.f;
    for (int n = slice * 16 + nl; n < N; n += 256) {
        const size_t idx = ((size_t)o * N + n) * 16 + b;
        float e = __expf(L[idx] - m);
        L[idx] = e;          // store unnormalized weights in place
        s += e;
    }
    __shared__ float red[256];
    red[threadIdx.x] = s;
    __syncthreads();
    for (int st = 128; st >= 16; st >>= 1) {
        if (threadIdx.x < st) red[threadIdx.x] += red[threadIdx.x + st];
        __syncthreads();
    }
    if (threadIdx.x < 16) P[((size_t)o * 16 + slice) * 16 + b] = red[threadIdx.x];
}

__global__ void k_zfin(const float* __restrict__ P, float* __restrict__ invZ) {
    const int o = blockIdx.x, b = threadIdx.x;
    if (b >= 16) return;
    float s = 0.f;
    for (int sl = 0; sl < 16; ++sl) s += P[((size_t)o * 16 + sl) * 16 + b];
    invZ[(size_t)o * 16 + b] = 1.0f / s;
}

// Hs[n][b] = sum_o H[o][n][b] * e[o][n][b] * invZ[o][b]
__global__ __launch_bounds__(256) void k_phaseC(const float* __restrict__ H,
                                                const float* __restrict__ Eatt,
                                                const float* __restrict__ invZ,
                                                float* __restrict__ Hs, int N) {
    __shared__ float sInv[FEAT * 16];
    for (int i = threadIdx.x; i < FEAT * 16; i += 256) sInv[i] = invZ[i];
    __syncthreads();
    int gid = blockIdx.x * 256 + threadIdx.x;
    int n = gid >> 4, b = gid & 15;
    if (n >= N) return;
    float acc = 0.f;
    for (int o = 0; o < FEAT; ++o) {
        const size_t idx = ((size_t)o * N + n) * 16 + b;
        acc = fmaf(H[idx], Eatt[idx] * sInv[o * 16 + b], acc);
    }
    Hs[(size_t)n * 16 + b] = acc;
}

// Y[b][n] = Hs[:, n, b] . out_w[n, 0:4] + ToD[b,n,:].out_w[n,4:28] + DoW*w[28] + out_b[n]
__global__ void k_final(const float* __restrict__ Hs,   // [4][N][16]
                        const float* __restrict__ ToD,  // [B][N][24]
                        const float* __restrict__ DoW,  // [B][N]
                        const float* __restrict__ out_w,// [N][29]
                        const float* __restrict__ out_b,
                        float* __restrict__ Y, int N) {
    int gid = blockIdx.x * 256 + threadIdx.x;
    if (gid >= BB * N) return;
    int n = gid % N, b = gid / N;
    const float* w = out_w + (size_t)n * 29;
    float acc = out_b[n];
    #pragma unroll
    for (int t = 0; t < TT; ++t)
        acc = fmaf(Hs[((size_t)t * N + n) * 16 + b], w[t], acc);
    const float* td = ToD + ((size_t)b * N + n) * 24;
    #pragma unroll
    for (int k = 0; k < 24; ++k) acc = fmaf(td[k], w[4 + k], acc);
    acc = fmaf(DoW[(size_t)b * N + n], w[28], acc);
    Y[(size_t)b * N + n] = acc;
}

// ---------------------------------------------------------------------
extern "C" void kernel_launch(void* const* d_in, const int* in_sizes, int n_in,
                              void* d_out, int out_size, void* d_ws, size_t ws_size,
                              hipStream_t stream) {
    const float* x      = (const float*)d_in[0];
    const float* t_vals = (const float*)d_in[1];
    const float* w_vals = (const float*)d_in[2];
    const float* ToD    = (const float*)d_in[3];
    const float* DoW    = (const float*)d_in[4];
    const float* att_w  = (const float*)d_in[5];
    const float* att_b  = (const float*)d_in[6];
    const float* out_w  = (const float*)d_in[7];
    const float* out_b  = (const float*)d_in[8];
    const int*   esrc   = (const int*)d_in[9];
    const int*   edst   = (const int*)d_in[10];
    float* Y = (float*)d_out;

    char* ws = (char*)d_ws;
    size_t cur = 0;
    auto alloc = [&](size_t bytes) -> char* {
        cur = (cur + 255) & ~(size_t)255;
        char* p = ws + cur;
        cur += bytes;
        return p;
    };
    const size_t NB = (size_t)NN * 16;        // one (N,B) slice
    int*   rowptrA = (int*)  alloc((NN + 1) * 4);
    int*   rowptrB = (int*)  alloc((NN + 1) * 4);
    int*   curA    = (int*)  alloc((size_t)NN * 4);   // doubles as cntA then cursorA
    int*   curB    = (int*)  alloc((size_t)NN * 4);
    int*   nbrA    = (int*)  alloc((size_t)EE * 4);
    int*   nbrB    = (int*)  alloc((size_t)EE * 4);
    float* wA      = (float*)alloc((size_t)EE * 4);
    float* wB      = (float*)alloc((size_t)EE * 4);
    float* tA      = (float*)alloc((size_t)TT * EE * 4);
    float* Hbuf    = (float*)alloc((size_t)FEAT * NB * 4);   // [76][N][16]
    float* Sbuf    = (float*)alloc((size_t)SFEAT * NB * 4);  // [15][N][16]
    float* Lbuf    = (float*)alloc((size_t)FEAT * NB * 4);   // [76][N][16]
    float* Pmax    = (float*)alloc((size_t)FEAT * 16 * 16 * 4);
    float* Mbuf    = (float*)alloc((size_t)FEAT * 16 * 4);
    float* Psum    = (float*)alloc((size_t)FEAT * 16 * 16 * 4);
    float* invZ    = (float*)alloc((size_t)FEAT * 16 * 4);
    float* Hs      = (float*)alloc((size_t)TT * NB * 4);     // [4][N][16]

    const int gN   = (NN + 255) / 256;
    const int gE   = (EE + 255) / 256;
    const int gNB  = (int)((NB + 255) / 256);
    const int gRow = (NN + 15) / 16;

    // ---- build CSR_A (by src) and CSR_B (by dst) once per launch ----
    k_zero_i<<<gN, 256, 0, stream>>>(curA, NN);
    k_zero_i<<<gN, 256, 0, stream>>>(curB, NN);
    k_hist<<<gE, 256, 0, stream>>>(esrc, edst, curA, curB, EE);
    k_scan<<<1, 1024, 0, stream>>>(curA, rowptrA, NN);
    k_scan<<<1, 1024, 0, stream>>>(curB, rowptrB, NN);
    k_copy_i<<<gN, 256, 0, stream>>>(rowptrA, curA, NN);
    k_copy_i<<<gN, 256, 0, stream>>>(rowptrB, curB, NN);
    k_scatter<<<gE, 256, 0, stream>>>(esrc, edst, w_vals, t_vals,
                                      curA, curB, nbrA, wA, tA, nbrB, wB, EE);

    // ---- per time-layer pipeline (sequential to bound workspace) ----
    for (int t = 0; t < TT; ++t) {
        // hop 0 slice (and status level-0 slice)
        k_transpose_x<<<gNB, 256, 0, stream>>>(x, Hbuf, Sbuf, t, NN);

        // demand diffusion: 75 hops of y[src] += t_val * x[dst]  (CSR_A gather)
        for (int hop = 1; hop <= HOPS; ++hop)
            k_spmv<<<gRow, 256, 0, stream>>>(rowptrA, nbrA, tA + (size_t)t * EE,
                                             Hbuf + (size_t)(hop - 1) * NB,
                                             Hbuf + (size_t)hop * NB, NN);

        // status diffusion: dual tree (W via CSR_B, W^T via CSR_A), 15 slices
        const int par[SFEAT] = {-1, 0, 0, 1, 1, 2, 2, 3, 3, 4, 4, 5, 5, 6, 6};
        for (int i = 1; i < SFEAT; ++i) {
            const bool isW = ((i & 1) == 1);  // odd index = W, even = W^T
            k_spmv<<<gRow, 256, 0, stream>>>(isW ? rowptrB : rowptrA,
                                             isW ? nbrB : nbrA,
                                             isW ? wB : wA,
                                             Sbuf + (size_t)par[i] * NB,
                                             Sbuf + (size_t)i * NB, NN);
        }

        // attention logits via WMMA (LDS-staged zero-padded f16 operands)
        k_wmma_att<<<(NN + 3) / 4, 128, 0, stream>>>(Sbuf, att_w, att_b, Lbuf, NN);

        // softmax over nodes + weighted feature contraction
        dim3 gr(FEAT, 16);
        k_maxpart<<<gr, 256, 0, stream>>>(Lbuf, Pmax, NN);
        k_maxfin<<<FEAT, 32, 0, stream>>>(Pmax, Mbuf);
        k_expsum<<<gr, 256, 0, stream>>>(Lbuf, Mbuf, Psum, NN);
        k_zfin<<<FEAT, 32, 0, stream>>>(Psum, invZ);
        k_phaseC<<<gNB, 256, 0, stream>>>(Hbuf, Lbuf, invZ, Hs + (size_t)t * NB, NN);
    }

    // final per-node linear head
    k_final<<<(BB * NN + 255) / 256, 256, 0, stream>>>(Hs, ToD, DoW, out_w, out_b, Y, NN);
}
